// NMT_17334488007145
// MI455X (gfx1250) — compile-verified, matrix operand order checked by reading, not run on hardware
//
#include <hip/hip_runtime.h>
#include <hip/hip_bf16.h>
#include <math.h>

// ---------------------------------------------------------------------------
// CDNA5 (gfx1250) NMT forward pass, round 5.
// GEMMs: v_wmma_f32_16x16x32_bf16; A panel (64 rows, shared by 8 waves)
// staged into LDS by the Tensor Data Mover (tensor_load_to_lds +
// s_wait_tensorcnt, double buffered); B streamed from L2-resident bf16
// weights. GEMM is templated on the K-chunk config so all LDS fragment
// addresses are base + constant immediate (no per-iteration address VALU).
// ---------------------------------------------------------------------------

typedef __attribute__((ext_vector_type(16))) __bf16 v16bf;
typedef __attribute__((ext_vector_type(8)))  __bf16 v8bf;
typedef __attribute__((ext_vector_type(8)))  float  v8f;
typedef __attribute__((ext_vector_type(4)))  unsigned int u32x4;
typedef __attribute__((ext_vector_type(8)))  int i32x8;
typedef __attribute__((ext_vector_type(4)))  int i32x4;

#define DEVI static __device__ __forceinline__

DEVI float sigmoidf_(float x) { return 1.0f / (1.0f + __expf(-x)); }

// fragment = [p[0..7], p[16..23]] (16B-aligned 16-bit WMMA operand layout)
DEVI v16bf load_frag(const __bf16* __restrict__ p)
{
    v8bf lo = *(const v8bf*)(p);
    v8bf hi = *(const v8bf*)(p + 16);
    return __builtin_shufflevector(lo, hi, 0, 1, 2, 3, 4, 5, 6, 7,
                                   8, 9, 10, 11, 12, 13, 14, 15);
}

// LDS fragment: lo at p, hi at p+32 bytes (16 bf16 further, same pad block)
DEVI v16bf load_frag_lds(const unsigned char* p)
{
    v8bf lo = *(const v8bf*)(p);
    v8bf hi = *(const v8bf*)(p + 32);
    return __builtin_shufflevector(lo, hi, 0, 1, 2, 3, 4, 5, 6, 7,
                                   8, 9, 10, 11, 12, 13, 14, 15);
}

// ---------------------------------------------------------------------------
// TDM: DMA a (64 x chunk) bf16 tile (row stride lda elements) from global
// into LDS at lds_addr, with pad_amount=4 DWORDs every 2^(pad_code+1) DWORDs.
// Descriptor layout per CDNA5 ISA 8.3-8.6 (2-D, data_size=2B, type=2).
// ---------------------------------------------------------------------------
DEVI void tdm_load_tile(const __bf16* gaddr, unsigned lds_addr, int lda_el,
                        int chunk, int pad_code)
{
    unsigned long long ga = (unsigned long long)(__UINTPTR_TYPE__)gaddr;
    u32x4 g0;
    g0[0] = 1u;                                   // count=1, user mode
    g0[1] = lds_addr;                             // LDS byte address
    g0[2] = (unsigned)(ga & 0xffffffffull);       // global_addr[31:0]
    g0[3] = (unsigned)((ga >> 32) & 0x1ffffffull) // global_addr[56:32]
          | (2u << 30);                           // type=2 ("image")
    i32x8 g1;
    g1[0] = (int)((1u << 16)                      // data_size: 2 bytes
                | (1u << 20)                      // pad_enable
                | ((unsigned)pad_code << 22)      // pad_interval
                | (3u << 25));                    // pad_amount = 4 DWORDs (16B)
    unsigned td0 = 1u << 20, td1 = 1u << 20;      // tensor dims: large (in-bounds)
    g1[1] = (int)((td0 & 0xffffu) << 16);
    g1[2] = (int)(((td0 >> 16) & 0xffffu) | ((td1 & 0xffffu) << 16));
    g1[3] = (int)(((td1 >> 16) & 0xffffu) | ((unsigned)chunk << 16)); // tile_dim0
    g1[4] = 64;                                   // tile_dim1=64 rows, tile_dim2=0
    g1[5] = (int)((unsigned)lda_el);              // tensor_dim0_stride[31:0]
    g1[6] = 0;                                    // stride0[47:32]=0, stride1 lo=0
    g1[7] = 0;
    i32x4 z4 = {0, 0, 0, 0};
    i32x8 z8 = {0, 0, 0, 0, 0, 0, 0, 0};
    __builtin_amdgcn_tensor_load_to_lds(g0, g1, z4, z4, z8, 0);
}

// ---------------------------------------------------------------------------
// WMMA GEMM: C(MxN,f32) = act([acc? C:0] + A(MxKp,bf16,lda) @ Bw(NpxKp,bf16)^T + bias)
// Block: 64 rows x (8 waves x 16 cols). A panel TDM-staged in LDS, double
// buffered per K-chunk. Requires M%64==0, Kp%CHUNK==0, buffers 16B aligned.
// Template constants make every LDS fragment address base+imm.
// ---------------------------------------------------------------------------
template <int CHUNK, int STRIDE, int PSHIFT, int PCODE>
__global__ void k_wmma_gemm_tdm(const __bf16* __restrict__ A, int lda,
                                const __bf16* __restrict__ Bw, int Kp,
                                float* __restrict__ C, int ldc,
                                __bf16* __restrict__ Cbf, int ldcbf,
                                const float* __restrict__ bias,
                                int M, int N, int acc_flag, int act_flag)
{
    __shared__ __align__(16) unsigned char smem[2][64 * STRIDE];

    const int lane = threadIdx.x & 31;
    const int wave = threadIdx.x >> 5;
    const int Nt   = (N + 15) >> 4;
    const int NtB  = (Nt + 7) >> 3;
    const int tg   = blockIdx.x / NtB;
    const int nb   = blockIdx.x % NtB;
    const int tn   = nb * 8 + wave;
    const bool active = tn < Nt;       // wave0 always active (nb*8 < Nt)
    const int rowg = tg << 6;
    const int col0 = tn << 4;

    const int kb   = (lane & 16) ? 8 : 0;   // lanes 16-31: K+8 / K+24 halves
    const int arow = lane & 15;
    const int bn   = col0 + arow;

    const __bf16* bp = Bw + (size_t)bn * (size_t)Kp + kb;
    const __bf16* apanel = A + (size_t)rowg * (size_t)lda;

    // defeat "never-written shared" folding; TDM overwrites all of it
    ((unsigned*)smem)[threadIdx.x] = 0u;

    v8f acc[4] = {v8f{}, v8f{}, v8f{}, v8f{}};

    const int nchunks = Kp / CHUNK;
    if (wave == 0)
        tdm_load_tile(apanel, (unsigned)(__UINTPTR_TYPE__)&smem[0][0],
                      lda, CHUNK, PCODE);

    for (int c = 0; c < nchunks; ++c) {
        if (wave == 0) {
            if (c + 1 < nchunks) {
                tdm_load_tile(apanel + (size_t)(c + 1) * CHUNK,
                              (unsigned)(__UINTPTR_TYPE__)&smem[(c + 1) & 1][0],
                              lda, CHUNK, PCODE);
                __builtin_amdgcn_s_wait_tensorcnt(1);   // chunk c complete
            } else {
                __builtin_amdgcn_s_wait_tensorcnt(0);
            }
        }
        __syncthreads();
        __asm__ __volatile__("" ::: "memory");

        if (active) {
            // per-chunk fragment base pointers: runtime part of the pad-aware
            // offset is just 2*kb (no carry into the pad block for our configs)
            const unsigned char* r0 = &smem[c & 1][0] + (size_t)arow * STRIDE + 2 * kb;
            const unsigned char* r1 = r0 + 16 * STRIDE;
            const unsigned char* r2 = r0 + 32 * STRIDE;
            const unsigned char* r3 = r0 + 48 * STRIDE;
            const __bf16* bpc = bp + c * CHUNK;
#pragma unroll
            for (int k0 = 0; k0 < CHUNK; k0 += 32) {
                const int off = 2 * k0 + 16 * (k0 >> PSHIFT);   // compile-time
                __builtin_prefetch((const void*)(bpc + k0 + 256), 0, 1);
                v16bf bf = load_frag(bpc + k0);
                v16bf a0 = load_frag_lds(r0 + off);
                v16bf a1 = load_frag_lds(r1 + off);
                v16bf a2 = load_frag_lds(r2 + off);
                v16bf a3 = load_frag_lds(r3 + off);
                acc[0] = __builtin_amdgcn_wmma_f32_16x16x32_bf16(false, a0, false, bf,
                                                                 (short)0, acc[0], false, false);
                acc[1] = __builtin_amdgcn_wmma_f32_16x16x32_bf16(false, a1, false, bf,
                                                                 (short)0, acc[1], false, false);
                acc[2] = __builtin_amdgcn_wmma_f32_16x16x32_bf16(false, a2, false, bf,
                                                                 (short)0, acc[2], false, false);
                acc[3] = __builtin_amdgcn_wmma_f32_16x16x32_bf16(false, a3, false, bf,
                                                                 (short)0, acc[3], false, false);
            }
        }
        __syncthreads();
    }

    if (!active) return;

    // epilogue: C/D layout: VGPR i -> row (i + 8*[lane>=16]), col = lane&15
    const int   mb  = (lane & 16) ? 8 : 0;
    const bool  nok = bn < N;
    float bv = 0.0f;
    if (bias && nok) bv = bias[bn];
#pragma unroll
    for (int mt = 0; mt < 4; ++mt) {
#pragma unroll
        for (int i = 0; i < 8; ++i) {
            int m = rowg + (mt << 4) + mb + i;
            float v = acc[mt][i] + bv;
            if (acc_flag && nok) v += C[(size_t)m * ldc + bn];
            if (act_flag) v = tanhf(v);
            if (C && nok) C[(size_t)m * ldc + bn] = v;
            if (Cbf) Cbf[(size_t)m * ldcbf + bn] = (__bf16)v;   // pad cols -> 0
        }
    }
}

// ---------------------------------------------------------------------------
// Weight conversion into zero-padded bf16 (Np x Kp), src is (N x K) f32
// ---------------------------------------------------------------------------
__global__ void k_cvt_w(const float* __restrict__ src, __bf16* __restrict__ dst,
                        int N, int K, int Kp, int total)
{
    int i = blockIdx.x * blockDim.x + threadIdx.x;
    if (i >= total) return;
    int n = i / Kp, k = i % Kp;
    dst[i] = (n < N && k < K) ? (__bf16)src[(size_t)n * K + k] : (__bf16)0.0f;
}

// transposed: src (K x N) f32 -> dst (Np x Kp): dst[n*Kp+k] = src[k*N+n]
__global__ void k_cvt_w_T(const float* __restrict__ src, __bf16* __restrict__ dst,
                          int N, int K, int Kp, int total)
{
    int i = blockIdx.x * blockDim.x + threadIdx.x;
    if (i >= total) return;
    int n = i / Kp, k = i % Kp;
    dst[i] = (n < N && k < K) ? (__bf16)src[(size_t)k * N + n] : (__bf16)0.0f;
}

__global__ void k_gather_rows_bf(const float* __restrict__ table, const int* __restrict__ ids,
                                 __bf16* __restrict__ out, int ntok, int E, int ldo)
{
    int i = blockIdx.x * blockDim.x + threadIdx.x;
    if (i >= ntok * ldo) return;
    int t = i / ldo, e = i % ldo;
    out[i] = (e < E) ? (__bf16)table[(size_t)ids[t] * E + e] : (__bf16)0.0f;
}

// prev[0] = init_emb broadcast; prev[t] = emb_de[trg[t-1]]  (f32, unpadded)
__global__ void k_gather_prev(const float* __restrict__ table, const int* __restrict__ trg,
                              const float* __restrict__ init_emb,
                              float* __restrict__ out, int Tn, int Bn, int E)
{
    int i = blockIdx.x * blockDim.x + threadIdx.x;
    if (i >= Tn * Bn * E) return;
    int e = i % E; int tb = i / E; int b = tb % Bn; int t = tb / Bn;
    out[i] = (t == 0) ? init_emb[e] : table[(size_t)trg[(t - 1) * Bn + b] * E + e];
}

__global__ void k_fill(float* __restrict__ dst, int n, float v)
{
    int i = blockIdx.x * blockDim.x + threadIdx.x;
    if (i < n) dst[i] = v;
}

__global__ void k_fill_bf(__bf16* __restrict__ dst, int n)
{
    int i = blockIdx.x * blockDim.x + threadIdx.x;
    if (i < n) dst[i] = (__bf16)0.0f;
}

__global__ void k_broadcast_row(const float* __restrict__ src, float* __restrict__ dst,
                                int rows, int N)
{
    int i = blockIdx.x * blockDim.x + threadIdx.x;
    if (i < rows * N) dst[i] = src[i % N];
}

// concat [A(Na) | Bm(Nb) | zero-pad]; optional f32 out (stride Na+Nb),
// optional padded bf16 out (stride ldo >= Na+Nb)
__global__ void k_concat2_dual(const float* __restrict__ A, int Na,
                               const float* __restrict__ Bm, int Nb,
                               float* __restrict__ outf,
                               __bf16* __restrict__ outb,
                               int rows, int ldo)
{
    int i = blockIdx.x * blockDim.x + threadIdx.x;
    if (i >= rows * ldo) return;
    int r = i / ldo, c = i % ldo;
    int N = Na + Nb;
    float v = (c < Na) ? A[(size_t)r * Na + c]
            : (c < N)  ? Bm[(size_t)r * Nb + (c - Na)]
                       : 0.0f;
    if (outf && c < N) outf[(size_t)r * N + c] = v;
    if (outb) outb[i] = (__bf16)v;
}

// gates (Bn x 4H) order [i,f,g,o]; updates c,h; writes bf16 h and optional copy
__global__ void k_lstm_cell(const float* __restrict__ gates, float* __restrict__ c,
                            float* __restrict__ h, __bf16* __restrict__ h_bf,
                            float* __restrict__ h_store, int Bn, int H)
{
    int i = blockIdx.x * blockDim.x + threadIdx.x;
    if (i >= Bn * H) return;
    int b = i / H, j = i % H;
    const float* g = gates + (size_t)b * 4 * H;
    float ig = sigmoidf_(g[j]);
    float fg = sigmoidf_(g[H + j]);
    float gg = tanhf(g[2 * H + j]);
    float og = sigmoidf_(g[3 * H + j]);
    float cn = fg * c[i] + ig * gg;
    float hn = og * tanhf(cn);
    c[i] = cn; h[i] = hn;
    h_bf[i] = (__bf16)hn;
    if (h_store) h_store[i] = hn;
}

// ---------------------------------------------------------------------------
// Attention + softmax kernels (wave32 reductions)
// ---------------------------------------------------------------------------
DEVI float block_reduce_sum(float v, float* sbuf)
{
    int lane = threadIdx.x & 31, w = threadIdx.x >> 5;
#pragma unroll
    for (int o = 16; o > 0; o >>= 1) v += __shfl_down(v, o, 32);
    if (lane == 0) sbuf[w] = v;
    __syncthreads();
    int nw = blockDim.x >> 5;
    v = (threadIdx.x < nw) ? sbuf[threadIdx.x] : 0.0f;
    if (w == 0) {
#pragma unroll
        for (int o = 16; o > 0; o >>= 1) v += __shfl_down(v, o, 32);
    }
    __syncthreads();
    return v; // valid in thread 0
}

DEVI float block_reduce_max(float v, float* sbuf)
{
    int lane = threadIdx.x & 31, w = threadIdx.x >> 5;
#pragma unroll
    for (int o = 16; o > 0; o >>= 1) v = fmaxf(v, __shfl_down(v, o, 32));
    if (lane == 0) sbuf[w] = v;
    __syncthreads();
    int nw = blockDim.x >> 5;
    v = (threadIdx.x < nw) ? sbuf[threadIdx.x] : -1e30f;
    if (w == 0) {
#pragma unroll
        for (int o = 16; o > 0; o >>= 1) v = fmaxf(v, __shfl_down(v, o, 32));
    }
    __syncthreads();
    return v;
}

__global__ void k_attn_scores(const float* __restrict__ ML, const float* __restrict__ dh,
                              float* __restrict__ scores, int Sn, int Bn, int D)
{
    __shared__ float sbuf[8];
    int s = blockIdx.x / Bn, b = blockIdx.x % Bn;
    const float* m = ML + ((size_t)s * Bn + b) * D;
    const float* h = dh + (size_t)b * D;
    float a = 0.0f;
    for (int d = threadIdx.x; d < D; d += blockDim.x) a += m[d] * h[d];
    a = block_reduce_sum(a, sbuf);
    if (threadIdx.x == 0) scores[b * Sn + s] = a;
}

__global__ void k_softmax_rows(const float* __restrict__ x, float* __restrict__ y, int Sn)
{
    __shared__ float sv[64];
    __shared__ float smax, ssum;
    int b = blockIdx.x, t = threadIdx.x;
    float v = (t < Sn) ? x[b * Sn + t] : -1e30f;
    sv[t] = v;
    __syncthreads();
    if (t == 0) {
        float mx = -1e30f;
        for (int i = 0; i < Sn; ++i) mx = fmaxf(mx, sv[i]);
        float sm = 0.0f;
        for (int i = 0; i < Sn; ++i) sm += __expf(sv[i] - mx);
        smax = mx; ssum = sm;
    }
    __syncthreads();
    if (t < Sn) y[b * Sn + t] = __expf(v - smax) / ssum;
}

__global__ void k_attn_ctx(const float* __restrict__ attn, const float* __restrict__ enc,
                           float* __restrict__ st, int Sn, int Bn, int D)
{
    int i = blockIdx.x * blockDim.x + threadIdx.x;
    if (i >= Bn * D) return;
    int b = i / D, d = i % D;
    float a = 0.0f;
    for (int s = 0; s < Sn; ++s) a += attn[b * Sn + s] * enc[((size_t)s * Bn + b) * D + d];
    st[i] = a;
}

__global__ void k_logsoftmax_rows(const float* __restrict__ x, float* __restrict__ y, int V)
{
    __shared__ float sbuf[8];
    __shared__ float smx, slse;
    int b = blockIdx.x;
    const float* xr = x + (size_t)b * V;
    float mx = -1e30f;
    for (int v = threadIdx.x; v < V; v += blockDim.x) mx = fmaxf(mx, xr[v]);
    mx = block_reduce_max(mx, sbuf);
    if (threadIdx.x == 0) smx = mx;
    __syncthreads();
    mx = smx;
    float sm = 0.0f;
    for (int v = threadIdx.x; v < V; v += blockDim.x) sm += __expf(xr[v] - mx);
    sm = block_reduce_sum(sm, sbuf);
    if (threadIdx.x == 0) slse = logf(sm);
    __syncthreads();
    float lse = slse;
    float* yr = y + (size_t)b * V;
    for (int v = threadIdx.x; v < V; v += blockDim.x) yr[v] = xr[v] - mx - lse;
}

// ---------------------------------------------------------------------------
// Host orchestration
// ---------------------------------------------------------------------------
static inline int ceil_div(int a, int b) { return (a + b - 1) / b; }
static inline int pad16(int x) { return (x + 15) & ~15; }
static inline int pad32(int x) { return (x + 31) & ~31; }

static inline void launch_gemm(const __bf16* A, int lda, const __bf16* Bw, int Kp,
                               float* C, int ldc, __bf16* Cbf, int ldcbf,
                               const float* bias, int M, int N,
                               int accf, int actf, hipStream_t s)
{
    int blocks = (M / 64) * ceil_div(ceil_div(N, 16), 8);
    if (Kp % 256 == 0)
        k_wmma_gemm_tdm<256, 528, 8, 6><<<blocks, 256, 0, s>>>(
            A, lda, Bw, Kp, C, ldc, Cbf, ldcbf, bias, M, N, accf, actf);
    else if (Kp % 192 == 0)
        k_wmma_gemm_tdm<192, 432, 6, 4><<<blocks, 256, 0, s>>>(
            A, lda, Bw, Kp, C, ldc, Cbf, ldcbf, bias, M, N, accf, actf);
    else
        k_wmma_gemm_tdm<64, 144, 6, 4><<<blocks, 256, 0, s>>>(
            A, lda, Bw, Kp, C, ldc, Cbf, ldcbf, bias, M, N, accf, actf);
}

extern "C" void kernel_launch(void* const* d_in, const int* in_sizes, int n_in,
                              void* d_out, int out_size, void* d_ws, size_t ws_size,
                              hipStream_t stream)
{
    (void)in_sizes; (void)n_in; (void)out_size; (void)ws_size;
    const int S = 50, T = 50, B = 64;
    const int EMB = 300, ENC_H = 512, DEC_H = 1024, CTX = 1024, TRG_V = 23262;
    const int EMB_P   = pad32(EMB);            // 320 -> chunk 64
    const int XK      = EMB + CTX;             // 1324
    const int XK_P    = pad32(XK);             // 1344 -> chunk 192
    const int CAT2    = CTX + DEC_H;           // 2048 -> chunk 256
    const int TRG_V_P = pad16(TRG_V);          // 23264

    const int*   src       = (const int*)  d_in[0];
    const int*   trg       = (const int*)  d_in[1];
    const float* emb_en    = (const float*)d_in[2];
    const float* emb_de    = (const float*)d_in[3];
    const float* enc_Wih_f = (const float*)d_in[4];
    const float* enc_Whh_f = (const float*)d_in[5];
    const float* enc_bih_f = (const float*)d_in[6];
    const float* enc_bhh_f = (const float*)d_in[7];
    const float* enc_Wih_b = (const float*)d_in[8];
    const float* enc_Whh_b = (const float*)d_in[9];
    const float* enc_bih_b = (const float*)d_in[10];
    const float* enc_bhh_b = (const float*)d_in[11];
    const float* dec_Wih   = (const float*)d_in[12];
    const float* dec_Whh   = (const float*)d_in[13];
    const float* dec_bih   = (const float*)d_in[14];
    const float* dec_bhh   = (const float*)d_in[15];
    const float* W_i       = (const float*)d_in[16];
    const float* W_o       = (const float*)d_in[17];
    const float* W_gen     = (const float*)d_in[18];
    const float* b_gen     = (const float*)d_in[19];
    const float* init_emb  = (const float*)d_in[20];
    const float* init_st   = (const float*)d_in[21];
    float*       out       = (float*)d_out;

    // ---- workspace bump allocator (256B aligned) ----
    char* base = (char*)d_ws;
    size_t off = 0;
    auto alloc = [&](size_t bytes) -> void* {
        off = (off + 255) & ~(size_t)255;
        void* p = base + off;
        off += bytes;
        return p;
    };

    // bf16 weights, zero-padded to (Np x Kp)
    __bf16* w_eih_f = (__bf16*)alloc((size_t)4 * ENC_H * EMB_P * 2);
    __bf16* w_ehh_f = (__bf16*)alloc((size_t)4 * ENC_H * ENC_H * 2);
    __bf16* w_eih_b = (__bf16*)alloc((size_t)4 * ENC_H * EMB_P * 2);
    __bf16* w_ehh_b = (__bf16*)alloc((size_t)4 * ENC_H * ENC_H * 2);
    __bf16* w_dih   = (__bf16*)alloc((size_t)4 * DEC_H * XK_P * 2);
    __bf16* w_dhh   = (__bf16*)alloc((size_t)4 * DEC_H * DEC_H * 2);
    __bf16* w_WiT   = (__bf16*)alloc((size_t)CTX * CTX * 2);          // W_i^T
    __bf16* w_Wo    = (__bf16*)alloc((size_t)DEC_H * CAT2 * 2);
    __bf16* w_Wgen  = (__bf16*)alloc((size_t)TRG_V_P * DEC_H * 2);

    // bf16 activations (GEMM A operands), padded strides
    __bf16* se_bf     = (__bf16*)alloc((size_t)S * B * EMB_P * 2);
    __bf16* hf_bf     = (__bf16*)alloc((size_t)B * ENC_H * 2);
    __bf16* hb_bf     = (__bf16*)alloc((size_t)B * ENC_H * 2);
    __bf16* dh_bf     = (__bf16*)alloc((size_t)B * DEC_H * 2);
    __bf16* xcat_bf   = (__bf16*)alloc((size_t)B * XK_P * 2);
    __bf16* cat2_bf   = (__bf16*)alloc((size_t)B * CAT2 * 2);
    __bf16* ct_bf     = (__bf16*)alloc((size_t)B * DEC_H * 2);
    __bf16* enccat_bf = (__bf16*)alloc((size_t)S * B * CTX * 2);

    // fp32 buffers
    float* prev_emb = (float*)alloc((size_t)T * B * EMB * 4);
    float* hs_f     = (float*)alloc((size_t)S * B * ENC_H * 4);
    float* hs_b     = (float*)alloc((size_t)S * B * ENC_H * 4);
    float* h_f      = (float*)alloc((size_t)B * ENC_H * 4);
    float* c_f      = (float*)alloc((size_t)B * ENC_H * 4);
    float* h_b      = (float*)alloc((size_t)B * ENC_H * 4);
    float* c_b      = (float*)alloc((size_t)B * ENC_H * 4);
    float* enc_cat  = (float*)alloc((size_t)S * B * CTX * 4);
    float* ML       = (float*)alloc((size_t)S * B * CTX * 4);
    float* d_h      = (float*)alloc((size_t)B * DEC_H * 4);
    float* d_c      = (float*)alloc((size_t)B * DEC_H * 4);
    float* s_t      = (float*)alloc((size_t)B * CTX * 4);
    float* gates    = (float*)alloc((size_t)B * 4 * DEC_H * 4);
    float* scores   = (float*)alloc((size_t)B * S * 4);
    float* attn     = (float*)alloc((size_t)B * S * 4);
    float* c_t      = (float*)alloc((size_t)B * DEC_H * 4);
    float* logits   = (float*)alloc((size_t)B * TRG_V * 4);

    const int TB = 256;
    auto cvt = [&](const float* s_, __bf16* d_, int N, int K, int Kp) {
        int total = pad16(N) * Kp;
        k_cvt_w<<<ceil_div(total, TB), TB, 0, stream>>>(s_, d_, N, K, Kp, total);
    };

    // ---- 1) weight conversion (padded bf16) ----
    cvt(enc_Wih_f, w_eih_f, 4 * ENC_H, EMB, EMB_P);
    cvt(enc_Whh_f, w_ehh_f, 4 * ENC_H, ENC_H, ENC_H);
    cvt(enc_Wih_b, w_eih_b, 4 * ENC_H, EMB, EMB_P);
    cvt(enc_Whh_b, w_ehh_b, 4 * ENC_H, ENC_H, ENC_H);
    cvt(dec_Wih,   w_dih,   4 * DEC_H, XK, XK_P);
    cvt(dec_Whh,   w_dhh,   4 * DEC_H, DEC_H, DEC_H);
    cvt(W_o,       w_Wo,    DEC_H, CAT2, CAT2);
    cvt(W_gen,     w_Wgen,  TRG_V, DEC_H, DEC_H);
    {
        int total = CTX * CTX;
        k_cvt_w_T<<<ceil_div(total, TB), TB, 0, stream>>>(W_i, w_WiT, CTX, CTX, CTX, total);
    }

    // ---- 2) embedding gathers ----
    k_gather_rows_bf<<<ceil_div(S * B * EMB_P, TB), TB, 0, stream>>>(
        emb_en, src, se_bf, S * B, EMB, EMB_P);
    k_gather_prev<<<ceil_div(T * B * EMB, TB), TB, 0, stream>>>(
        emb_de, trg, init_emb, prev_emb, T, B, EMB);

    // ---- 3) encoder (fwd + bwd LSTM) ----
    k_fill<<<ceil_div(B * ENC_H, TB), TB, 0, stream>>>(h_f, B * ENC_H, 0.0f);
    k_fill<<<ceil_div(B * ENC_H, TB), TB, 0, stream>>>(c_f, B * ENC_H, 0.0f);
    k_fill<<<ceil_div(B * ENC_H, TB), TB, 0, stream>>>(h_b, B * ENC_H, 0.0f);
    k_fill<<<ceil_div(B * ENC_H, TB), TB, 0, stream>>>(c_b, B * ENC_H, 0.0f);
    k_fill_bf<<<ceil_div(B * ENC_H, TB), TB, 0, stream>>>(hf_bf, B * ENC_H);
    k_fill_bf<<<ceil_div(B * ENC_H, TB), TB, 0, stream>>>(hb_bf, B * ENC_H);

    for (int t = 0; t < S; ++t) {
        // forward direction
        launch_gemm(se_bf + (size_t)t * B * EMB_P, EMB_P, w_eih_f, EMB_P,
                    gates, 4 * ENC_H, nullptr, 0, enc_bih_f, B, 4 * ENC_H, 0, 0, stream);
        launch_gemm(hf_bf, ENC_H, w_ehh_f, ENC_H,
                    gates, 4 * ENC_H, nullptr, 0, enc_bhh_f, B, 4 * ENC_H, 1, 0, stream);
        k_lstm_cell<<<ceil_div(B * ENC_H, TB), TB, 0, stream>>>(
            gates, c_f, h_f, hf_bf, hs_f + (size_t)t * B * ENC_H, B, ENC_H);
        // backward direction (reversed input, stores reversed)
        int rt = S - 1 - t;
        launch_gemm(se_bf + (size_t)rt * B * EMB_P, EMB_P, w_eih_b, EMB_P,
                    gates, 4 * ENC_H, nullptr, 0, enc_bih_b, B, 4 * ENC_H, 0, 0, stream);
        launch_gemm(hb_bf, ENC_H, w_ehh_b, ENC_H,
                    gates, 4 * ENC_H, nullptr, 0, enc_bhh_b, B, 4 * ENC_H, 1, 0, stream);
        k_lstm_cell<<<ceil_div(B * ENC_H, TB), TB, 0, stream>>>(
            gates, c_b, h_b, hb_bf, hs_b + (size_t)rt * B * ENC_H, B, ENC_H);
    }

    // ---- 4) enc concat (f32 + bf16), decoder init, ML = enc @ W_i ----
    k_concat2_dual<<<ceil_div(S * B * CTX, TB), TB, 0, stream>>>(
        hs_f, ENC_H, hs_b, ENC_H, enc_cat, enccat_bf, S * B, CTX);
    k_concat2_dual<<<ceil_div(B * DEC_H, TB), TB, 0, stream>>>(
        h_f, ENC_H, h_b, ENC_H, d_h, dh_bf, B, DEC_H);
    k_concat2_dual<<<ceil_div(B * DEC_H, TB), TB, 0, stream>>>(
        c_f, ENC_H, c_b, ENC_H, d_c, nullptr, B, DEC_H);
    launch_gemm(enccat_bf, CTX, w_WiT, CTX, ML, CTX, nullptr, 0, nullptr,
                S * B, CTX, 0, 0, stream);
    k_broadcast_row<<<ceil_div(B * CTX, TB), TB, 0, stream>>>(init_st, s_t, B, CTX);

    // ---- 5) decoder ----
    for (int t = 0; t < T; ++t) {
        k_concat2_dual<<<ceil_div(B * XK_P, TB), TB, 0, stream>>>(
            s_t, CTX, prev_emb + (size_t)t * B * EMB, EMB, nullptr, xcat_bf, B, XK_P);
        launch_gemm(xcat_bf, XK_P, w_dih, XK_P, gates, 4 * DEC_H, nullptr, 0,
                    dec_bih, B, 4 * DEC_H, 0, 0, stream);
        launch_gemm(dh_bf, DEC_H, w_dhh, DEC_H, gates, 4 * DEC_H, nullptr, 0,
                    dec_bhh, B, 4 * DEC_H, 1, 0, stream);
        k_lstm_cell<<<ceil_div(B * DEC_H, TB), TB, 0, stream>>>(
            gates, d_c, d_h, dh_bf, nullptr, B, DEC_H);

        k_attn_scores<<<S * B, 128, 0, stream>>>(ML, d_h, scores, S, B, CTX);
        k_softmax_rows<<<B, 64, 0, stream>>>(scores, attn, S);
        k_attn_ctx<<<ceil_div(B * CTX, TB), TB, 0, stream>>>(attn, enc_cat, s_t, S, B, CTX);

        k_concat2_dual<<<ceil_div(B * CAT2, TB), TB, 0, stream>>>(
            s_t, CTX, d_h, DEC_H, nullptr, cat2_bf, B, CAT2);
        launch_gemm(cat2_bf, CAT2, w_Wo, CAT2, c_t, DEC_H, ct_bf, DEC_H,
                    nullptr, B, DEC_H, 0, 1, stream);             // tanh + bf16 out
        launch_gemm(ct_bf, DEC_H, w_Wgen, DEC_H, logits, TRG_V, nullptr, 0,
                    b_gen, B, TRG_V, 0, 0, stream);
        k_logsoftmax_rows<<<B, 256, 0, stream>>>(logits, out + (size_t)t * B * TRG_V, TRG_V);
    }
}
